// Seq2SeqModel_33234456936870
// MI455X (gfx1250) — matrix operand order repaired and checked by reading, not compiled, not run on hardware
//
#include <hip/hip_runtime.h>

#define DIN   69
#define DPAD  96
#define HID   2048
#define BATCH 512
#define SRCL  100
#define TGTL  50

#define BM 128
#define BN 32
#define CH 32

#ifndef TRY_ASYNC
#define TRY_ASYNC 1
#endif

#if TRY_ASYNC && defined(__AMDGCN__) && \
    __has_builtin(__builtin_amdgcn_global_load_async_to_lds_b128) && \
    __has_builtin(__builtin_amdgcn_s_wait_asynccnt)
#define USE_ASYNC 1
#else
#define USE_ASYNC 0
#endif

typedef __attribute__((ext_vector_type(16))) __bf16 v16bf;
typedef __attribute__((ext_vector_type(8)))  float  v8f;
typedef __attribute__((__vector_size__(4 * sizeof(int)))) int v4i_t;

#define WMMA(A, Bm, C) \
  __builtin_amdgcn_wmma_f32_16x16x32_bf16(false, (A), false, (Bm), (short)0, (C), false, false)

static __device__ __forceinline__ unsigned short f2bf(float f) {
  union { float f; unsigned u; } x; x.f = f;
  return (unsigned short)((x.u + 0x7FFFu + ((x.u >> 16) & 1u)) >> 16);
}

// 16-byte global -> LDS copy (per-lane). Async path uses the CDNA5 async
// global->LDS instruction tracked with ASYNCcnt; fallback is load + ds_store.
static __device__ __forceinline__ void copy16(const unsigned short* g, unsigned short* l) {
#if USE_ASYNC
  __builtin_amdgcn_global_load_async_to_lds_b128(
      (__attribute__((address_space(1))) v4i_t*)g,
      (__attribute__((address_space(3))) v4i_t*)l, 0, 0);
#else
  *(uint4*)(void*)l = *(const uint4*)(const void*)g;
#endif
}

static __device__ __forceinline__ void waitcp() {
#if USE_ASYNC
  __builtin_amdgcn_s_wait_asynccnt(0);
#endif
}

// A-matrix fragment (16-bit 16x32 ISA layout): per lane, K groups {kb..kb+7, kb+16..kb+23}
static __device__ __forceinline__ v16bf ldfragA(const unsigned short* p) {
  union { uint4 u[2]; v16bf v; } f;
  f.u[0] = *(const uint4*)(const void*)(p);
  f.u[1] = *(const uint4*)(const void*)(p + 16);
  return f.v;
}
// B-matrix fragment: per lane (column), K groups {kc..kc+7, kc+16..kc+23}
static __device__ __forceinline__ v16bf ldfragB(const unsigned short* p) {
  union { uint4 u[2]; v16bf v; } f;
  f.u[0] = *(const uint4*)(const void*)(p);
  f.u[1] = *(const uint4*)(const void*)(p + 16);
  return f.v;
}

// ---------------- one-time prep kernels ----------------

__global__ void k_cvt(const float* __restrict__ src, unsigned short* __restrict__ dst,
                      int rows_in, int kin, int kout, long rows_out) {
  long total = rows_out * (long)kout;
  for (long i = blockIdx.x * (long)blockDim.x + threadIdx.x; i < total;
       i += (long)gridDim.x * blockDim.x) {
    long r = i / kout; int k = (int)(i - r * kout);
    float v = (r < rows_in && k < kin) ? src[r * kin + k] : 0.f;
    dst[i] = f2bf(v);
  }
}

// encoder_inputs [B][SRCL][DIN] fp32 -> [SRCL][B][DPAD] bf16 (zero-padded)
__global__ void k_cvt_xenc(const float* __restrict__ enc, unsigned short* __restrict__ dst) {
  long total = (long)SRCL * BATCH * DPAD;
  for (long i = blockIdx.x * (long)blockDim.x + threadIdx.x; i < total;
       i += (long)gridDim.x * blockDim.x) {
    int k = (int)(i % DPAD); long rb = i / DPAD;
    int b = (int)(rb % BATCH); int t = (int)(rb / BATCH);
    float v = (k < DIN) ? enc[((long)b * SRCL + t) * DIN + k] : 0.f;
    dst[i] = f2bf(v);
  }
}

// zero h0 (fp32 + bf16), load decoder_inputs[:,0,:] into x buffers
__global__ void k_init(float* __restrict__ hf, unsigned short* __restrict__ hb,
                       const float* __restrict__ dec, float* __restrict__ xf,
                       unsigned short* __restrict__ xbf) {
  long nh = (long)BATCH * HID;
  for (long i = blockIdx.x * (long)blockDim.x + threadIdx.x; i < nh;
       i += (long)gridDim.x * blockDim.x) { hf[i] = 0.f; hb[i] = 0; }
  long nx = (long)BATCH * DPAD;
  for (long i = blockIdx.x * (long)blockDim.x + threadIdx.x; i < nx;
       i += (long)gridDim.x * blockDim.x) {
    int k = (int)(i % DPAD); long b = i / DPAD;
    float v = (k < DIN) ? dec[b * (TGTL * DIN) + k] : 0.f;
    xf[i] = v; xbf[i] = f2bf(v);
  }
}

// ---------------- fused GRU step (3 gates + elementwise update) ----------------
// grid (BATCH/BM, HID/BN), 256 threads (8 waves). Wave wv: wm=wv&3 (M offset),
// wn=wv>>2 (N offset). Each wave owns a 32x16 output region for each of 3 gates.
// K loop is split in two phases (x: 3 chunks -> accNi, h: 64 chunks -> accNh) so
// every WMMA has a fixed accumulator (no in-loop accumulator select / v_mov storms).

__global__ __launch_bounds__(256) void k_gru_step(
    const unsigned short* __restrict__ xsrc,   // [BATCH][DPAD] bf16
    const unsigned short* __restrict__ wih,    // [3H][DPAD] bf16
    const unsigned short* __restrict__ whh,    // [3H][HID]  bf16
    const unsigned short* __restrict__ hbf_in, // [BATCH][HID] bf16
    const float* __restrict__ hf_in,           // [BATCH][HID] fp32
    float* __restrict__ hf_out,
    unsigned short* __restrict__ hbf_out,
    const float* __restrict__ b_ih,
    const float* __restrict__ b_hh) {
  __shared__ alignas(16) unsigned short sA[2][BM * CH];
  __shared__ alignas(16) unsigned short sB[2][3 * BN * CH];

  const int tid = threadIdx.x;
  const int m0 = blockIdx.x * BM;
  const int n0 = blockIdx.y * BN;
  const int NCX = 3;            // x chunks (K = 96)
  const int NCH = HID / CH;     // h chunks (K = 2048)
  const int NC = NCX + NCH;

  const int lane = tid & 31;
  const int wv = tid >> 5;
  const int wm = wv & 3;
  const int wn = wv >> 2;
  const int lr = lane & 15;
  const int kb = (lane >> 4) * 8;
  const int kc = (lane >> 4) * 8;

  auto issue = [&](int bufi, int c) {
    const unsigned short* A; long sa; const unsigned short* W; long sw;
    if (c < NCX) { A = xsrc + (long)m0 * DPAD + c * CH; sa = DPAD; W = wih + c * CH; sw = DPAD; }
    else { int kk = (c - NCX) * CH; A = hbf_in + (long)m0 * HID + kk; sa = HID; W = whh + kk; sw = HID; }
    for (int s = tid; s < (BM * CH) / 8; s += 256) {           // 512 x 16B
      int m = s >> 2, part = s & 3;
      copy16(A + m * sa + part * 8, &sA[bufi][m * CH + part * 8]);
    }
    for (int s = tid; s < (3 * BN * CH) / 8; s += 256) {       // 384 x 16B
      int rowi = s >> 2, part = s & 3;
      int g = rowi >> 5, n = rowi & 31;
      copy16(W + (long)(n0 + n + g * HID) * sw + part * 8, &sB[bufi][rowi * CH + part * 8]);
    }
  };

  const v8f zero = {0.f, 0.f, 0.f, 0.f, 0.f, 0.f, 0.f, 0.f};
  v8f accR[2]  = {zero, zero};
  v8f accZ[2]  = {zero, zero};
  v8f accNi[2] = {zero, zero};
  v8f accNh[2] = {zero, zero};

  const int nrow = wn * 16 + lr;
  const int aoff0 = (wm * 32 + lr) * CH + kb;
  const int aoff1 = (wm * 32 + 16 + lr) * CH + kb;
  const int boff0 = (0 * BN + nrow) * CH + kc;
  const int boff1 = (1 * BN + nrow) * CH + kc;
  const int boff2 = (2 * BN + nrow) * CH + kc;

  issue(0, 0);
  waitcp();
  __syncthreads();

  int cc = 0;
  // ---- phase 1: x @ W_ih^T (3 chunks), n-gate goes to accNi ----
  for (int c = 0; c < NCX; ++c, ++cc) {
    const int buf = cc & 1;
    issue(buf ^ 1, cc + 1);  // cc+1 == NCX on last iter -> first h chunk

    const unsigned short* Ab = &sA[buf][0];
    const unsigned short* Bb = &sB[buf][0];
    v16bf a0 = ldfragA(Ab + aoff0);
    v16bf a1 = ldfragA(Ab + aoff1);
    v16bf b0 = ldfragB(Bb + boff0);
    v16bf b1 = ldfragB(Bb + boff1);
    v16bf b2 = ldfragB(Bb + boff2);

    accR[0] = WMMA(a0, b0, accR[0]);
    accR[1] = WMMA(a1, b0, accR[1]);
    accZ[0] = WMMA(a0, b1, accZ[0]);
    accZ[1] = WMMA(a1, b1, accZ[1]);
    accNi[0] = WMMA(a0, b2, accNi[0]);
    accNi[1] = WMMA(a1, b2, accNi[1]);

    waitcp();
    __syncthreads();
  }
  // ---- phase 2: h @ W_hh^T (64 chunks), n-gate goes to accNh ----
  for (int c = 0; c < NCH; ++c, ++cc) {
    const int buf = cc & 1;
    if (cc + 1 < NC) issue(buf ^ 1, cc + 1);

    const unsigned short* Ab = &sA[buf][0];
    const unsigned short* Bb = &sB[buf][0];
    v16bf a0 = ldfragA(Ab + aoff0);
    v16bf a1 = ldfragA(Ab + aoff1);
    v16bf b0 = ldfragB(Bb + boff0);
    v16bf b1 = ldfragB(Bb + boff1);
    v16bf b2 = ldfragB(Bb + boff2);

    accR[0] = WMMA(a0, b0, accR[0]);
    accR[1] = WMMA(a1, b0, accR[1]);
    accZ[0] = WMMA(a0, b1, accZ[0]);
    accZ[1] = WMMA(a1, b1, accZ[1]);
    accNh[0] = WMMA(a0, b2, accNh[0]);
    accNh[1] = WMMA(a1, b2, accNh[1]);

    if (cc + 1 < NC) waitcp();
    __syncthreads();
  }

  const int j = n0 + wn * 16 + lr;
  const float bR  = b_ih[j] + b_hh[j];
  const float bZ  = b_ih[j + HID] + b_hh[j + HID];
  const float bNi = b_ih[j + 2 * HID];
  const float bNh = b_hh[j + 2 * HID];
  const int rsel = (lane >> 4) * 8;

  for (int mi = 0; mi < 2; ++mi) {
    const int rowb = m0 + wm * 32 + mi * 16 + rsel;
    for (int v = 0; v < 8; ++v) {
      const int row = rowb + v;
      float r = 1.f / (1.f + __expf(-(accR[mi][v] + bR)));
      float z = 1.f / (1.f + __expf(-(accZ[mi][v] + bZ)));
      float n = tanhf(accNi[mi][v] + bNi + r * (accNh[mi][v] + bNh));
      float hnew = (1.f - z) * n + z * hf_in[(long)row * HID + j];
      hf_out[(long)row * HID + j] = hnew;
      hbf_out[(long)row * HID + j] = f2bf(hnew);
    }
  }
}

// ---------------- decoder output: out = x + h_new @ fc1^T + b; feeds next x ----------------
// grid (BATCH/16, 80/16), 1 wave. Small K=2048 WMMA GEMM straight from global (L2-resident).

__global__ __launch_bounds__(32) void k_out_step(
    const unsigned short* __restrict__ hbf,    // new h, bf16
    const unsigned short* __restrict__ fc1bf,  // [80][HID] bf16 (rows >=69 zero)
    const float* __restrict__ fc1b,
    float* __restrict__ xf,                    // [BATCH][DPAD] fp32 (in/out)
    unsigned short* __restrict__ xbf,          // [BATCH][DPAD] bf16 (out)
    float* __restrict__ out, int t) {
  const int lane = threadIdx.x;
  const int lr = lane & 15;
  const int kb = (lane >> 4) * 8;
  const int kc = (lane >> 4) * 8;
  const long mrow = blockIdx.x * 16 + lr;
  const long ncol = blockIdx.y * 16 + lr;

  v8f acc = {0.f, 0.f, 0.f, 0.f, 0.f, 0.f, 0.f, 0.f};
  for (int kk = 0; kk < HID; kk += CH) {
    v16bf a = ldfragA(hbf + mrow * HID + kk + kb);
    v16bf b = ldfragB(fc1bf + ncol * HID + kk + kc);
    acc = WMMA(a, b, acc);
  }

  const int j = blockIdx.y * 16 + lr;
  if (j < DIN) {
    const float fb = fc1b[j];
    for (int v = 0; v < 8; ++v) {
      const int bI = blockIdx.x * 16 + (lane >> 4) * 8 + v;
      float o = xf[(long)bI * DPAD + j] + acc[v] + fb;
      out[(long)bI * (TGTL * DIN) + t * DIN + j] = o;
      xf[(long)bI * DPAD + j] = o;
      xbf[(long)bI * DPAD + j] = f2bf(o);
    }
  }
}

// ---------------- host launcher ----------------

extern "C" void kernel_launch(void* const* d_in, const int* in_sizes, int n_in,
                              void* d_out, int out_size, void* d_ws, size_t ws_size,
                              hipStream_t stream) {
  (void)in_sizes; (void)n_in; (void)out_size; (void)ws_size;
  const float* enc  = (const float*)d_in[0];
  const float* dec  = (const float*)d_in[1];
  const float* W_ih = (const float*)d_in[2];
  const float* W_hh = (const float*)d_in[3];
  const float* b_ih = (const float*)d_in[4];
  const float* b_hh = (const float*)d_in[5];
  const float* fc1w = (const float*)d_in[6];
  const float* fc1b = (const float*)d_in[7];
  float* out = (float*)d_out;

  char* ws = (char*)d_ws;
  size_t off = 0;
  auto alloc = [&](size_t bytes) -> char* {
    char* p = ws + off; off += (bytes + 255) & ~(size_t)255; return p;
  };
  unsigned short* whh_bf  = (unsigned short*)alloc((size_t)3 * HID * HID * 2);
  unsigned short* wih_bf  = (unsigned short*)alloc((size_t)3 * HID * DPAD * 2);
  unsigned short* fc1_bf  = (unsigned short*)alloc((size_t)80 * HID * 2);
  unsigned short* xenc_bf = (unsigned short*)alloc((size_t)SRCL * BATCH * DPAD * 2);
  float*          hf[2];
  unsigned short* hb[2];
  hf[0] = (float*)alloc((size_t)BATCH * HID * 4);
  hf[1] = (float*)alloc((size_t)BATCH * HID * 4);
  hb[0] = (unsigned short*)alloc((size_t)BATCH * HID * 2);
  hb[1] = (unsigned short*)alloc((size_t)BATCH * HID * 2);
  float*          xf  = (float*)alloc((size_t)BATCH * DPAD * 4);
  unsigned short* xbf = (unsigned short*)alloc((size_t)BATCH * DPAD * 2);

  // one-time per-call prep
  k_cvt<<<2048, 256, 0, stream>>>(W_hh, whh_bf, 3 * HID, HID, HID, 3 * HID);
  k_cvt<<<512,  256, 0, stream>>>(W_ih, wih_bf, 3 * HID, DIN, DPAD, 3 * HID);
  k_cvt<<<128,  256, 0, stream>>>(fc1w, fc1_bf, DIN, HID, HID, 80);
  k_cvt_xenc<<<2048, 256, 0, stream>>>(enc, xenc_bf);
  k_init<<<1024, 256, 0, stream>>>(hf[0], hb[0], dec, xf, xbf);

  const dim3 ggrid(BATCH / BM, HID / BN);
  int s = 0;
  // encoder: SRC_LEN-1 = 99 steps
  for (int t = 0; t < SRCL - 1; ++t, ++s) {
    k_gru_step<<<ggrid, 256, 0, stream>>>(
        xenc_bf + (size_t)t * BATCH * DPAD, wih_bf, whh_bf,
        hb[s & 1], hf[s & 1], hf[(s + 1) & 1], hb[(s + 1) & 1], b_ih, b_hh);
  }
  // decoder: 50 autoregressive steps
  for (int t = 0; t < TGTL; ++t, ++s) {
    k_gru_step<<<ggrid, 256, 0, stream>>>(
        xbf, wih_bf, whh_bf,
        hb[s & 1], hf[s & 1], hf[(s + 1) & 1], hb[(s + 1) & 1], b_ih, b_hh);
    k_out_step<<<dim3(BATCH / 16, 80 / 16), 32, 0, stream>>>(
        hb[(s + 1) & 1], fc1_bf, fc1b, xf, xbf, out, t);
  }
}
